// MultiHeadedAttention_31731218382885
// MI455X (gfx1250) — matrix operand order repaired and checked by reading, not compiled, not run on hardware
//
#include <hip/hip_runtime.h>

// ---------------------------------------------------------------------------
// Structured multi-head attention for MI455X (gfx1250).
// bf16 WMMA pipeline + Tensor Data Mover (TDM) double-buffered staging of the
// 252 MB structure tensor into LDS (tracked with TENSORcnt).
// BS=8, L=512, D_MODEL=256, N_HEAD=8, D_HEAD=32, D_STRUCT=30, ALPHA=BETA=1.
// ---------------------------------------------------------------------------

#define BS 8
#define LQ 512
#define DM 256
#define NH 8
#define DH 32
#define DS 30
#define ALPHA_C 1.0f
#define BETA_C 1.0f

typedef __attribute__((ext_vector_type(16))) __bf16    v16bf;
typedef __attribute__((ext_vector_type(8)))  float     v8f;
typedef __attribute__((ext_vector_type(4)))  unsigned  u32x4;
typedef __attribute__((ext_vector_type(8)))  unsigned  u32x8;

union BF16x16 { unsigned short u[16]; v16bf v; };

__device__ __forceinline__ unsigned short f2bf(float f) {
  __bf16 h = (__bf16)f;                       // native cvt on gfx1250 (RNE)
  return __builtin_bit_cast(unsigned short, h);
}
__device__ __forceinline__ float bf2f(unsigned short b) {
  return (float)__builtin_bit_cast(__bf16, b);
}
__device__ __forceinline__ v8f wmma_bf16(v16bf a, v16bf b, v8f c) {
  // D = A(16x32) * B(32x16) + C, f32 accumulate
  return __builtin_amdgcn_wmma_f32_16x16x32_bf16(false, a, false, b,
                                                 (short)0, c, false, false);
}
// A-fragment (16-bit, 16x32) per-lane K index for element e (contiguous: k = 16*hv + e)
__device__ __forceinline__ int a_k(int e, int hv) {
  return e + 16 * hv;
}

// ---------------------------------------------------------------------------
// TDM: DMA one structure tile [16 q][16 k][30 f32] from global into LDS.
// 3D tile: x = 30 elems (contig), y = 16 lines stride 30, z = 16 lines
// stride 512*30.  D# per CDNA5 ISA 8.3-8.6; data_size=4B, type=2 (image).
// Issued by one wave; completion tracked with TENSORcnt.
// ---------------------------------------------------------------------------
__device__ __forceinline__ void tdm_load_tile(const float* gptr, void* ldsPtr) {
  const unsigned long long ga = (unsigned long long)(size_t)gptr;
  const unsigned lds = (unsigned)(size_t)ldsPtr;     // LDS byte offset
  u32x4 g0; u32x8 g1; u32x4 g2; u32x4 g3;
  g0[0] = 1u;                                        // count=1, user mode
  g0[1] = lds;                                       // lds_addr
  g0[2] = (unsigned)(ga & 0xffffffffu);              // global_addr[31:0]
  g0[3] = (unsigned)((ga >> 32) & 0x01ffffffu) | (2u << 30);  // addr[56:32]|type=2
  g1[0] = (2u << 16);                                // data_size=2 (4 bytes)
  g1[1] = ((unsigned)DS << 16);                      // tensor_dim0 = 30 (lo16)
  g1[2] = 0u;                                        // dim0 hi | tensor_dim1 lo16 (1<<20 -> 0)
  g1[3] = 0x10u | ((unsigned)DS << 16);              // tensor_dim1 hi | tile_dim0=30
  g1[4] = 16u | (16u << 16);                         // tile_dim1=16 | tile_dim2=16
  g1[5] = (unsigned)DS;                              // tensor_dim0_stride = 30
  g1[6] = (unsigned)(LQ * DS) << 16;                 // dim0_stride hi | dim1_stride lo16 = 15360
  g1[7] = 0u;                                        // dim1_stride hi
  g2[0] = (1u << 20);                                // tensor_dim2 (large)
  g2[1] = 0u; g2[2] = 0u; g2[3] = 0u;                // no dim3 / no iterate
  g3[0] = 0u; g3[1] = 0u; g3[2] = 0u; g3[3] = 0u;
  asm volatile("tensor_load_to_lds %0, %1, %2, %3"
               :: "s"(g0), "s"(g1), "s"(g2), "s"(g3)
               : "memory");
}

// ---------------------------------------------------------------------------
// Kernel 1: q/k/v projections.  q=(x@wq+bq)/sqrt(32), k=x@wk+bk, v=x@wv+bv.
// Output bf16 [b*512, 256].  One wave per 16x16 output tile, K=256 in 8 steps.
// ---------------------------------------------------------------------------
__global__ __launch_bounds__(256) void proj_qkv_kernel(
    const float* __restrict__ query, const float* __restrict__ key,
    const float* __restrict__ value,
    const float* __restrict__ wq, const float* __restrict__ bq,
    const float* __restrict__ wk, const float* __restrict__ bk,
    const float* __restrict__ wv, const float* __restrict__ bv,
    unsigned short* __restrict__ qb, unsigned short* __restrict__ kb,
    unsigned short* __restrict__ vb)
{
  const int tid  = threadIdx.x;
  const int lane = tid & 31;
  const int wave = tid >> 5;
  const int lm   = lane & 15;
  const int hv   = lane >> 4;

  const int tileId = blockIdx.x * 8 + wave;      // 12288 tiles total
  const int mat    = tileId >> 12;               // 0:q 1:k 2:v
  const int rem    = tileId & 4095;
  const int m0     = (rem >> 4) * 16;
  const int n0     = (rem & 15) * 16;

  const float* src; const float* w; const float* bias;
  unsigned short* dst; float scale;
  if (mat == 0)      { src = query; w = wq; bias = bq; dst = qb; scale = 0.17677669529663687f; }
  else if (mat == 1) { src = key;   w = wk; bias = bk; dst = kb; scale = 1.0f; }
  else               { src = value; w = wv; bias = bv; dst = vb; scale = 1.0f; }

  v8f acc = {};
  for (int kk = 0; kk < 8; ++kk) {
    const int kb0 = kk * 32;
    BF16x16 A, B;
    const float* arow = src + (size_t)(m0 + lm) * DM + kb0;
#pragma unroll
    for (int e = 0; e < 16; ++e) A.u[e] = f2bf(arow[a_k(e, hv)]);
    const int bn = n0 + lm;
#pragma unroll
    for (int e = 0; e < 16; ++e) {
      const int kr = kb0 + 16 * hv + e;
      B.u[e] = f2bf(w[(size_t)kr * DM + bn]);
    }
    acc = wmma_bf16(A.v, B.v, acc);
  }
#pragma unroll
  for (int r = 0; r < 8; ++r) {
    const int m = r + 8 * hv;
    const int n = n0 + lm;
    dst[(size_t)(m0 + m) * DM + n] = f2bf((acc[r] + bias[n]) * scale);
  }
}

// ---------------------------------------------------------------------------
// Kernel 2: fused attention.  One block = (batch b, 16-query tile), 8 waves,
// wave h owns head h.  LDS (dynamic, 217088 B):
//   sScore  [8][16][512] bf16  131072 B   scores -> exp(s-max)
//   sQ      [16][256]    bf16    8192 B   projected q rows
//   sSkt    [256][32]    bf16   16384 B   sk / sv tile (WMMA output)
//   sStage0 [256][30]    f32    30720 B   TDM target, buffer 0
//   sStage1 [256][30]    f32    30720 B   TDM target, buffer 1
//   sCtx    [16][256]    bf16   (overlaps sStage0, phase 4 only)
// ---------------------------------------------------------------------------
__global__ __launch_bounds__(256) void attn_kernel(
    const unsigned short* __restrict__ qb, const unsigned short* __restrict__ kbuf,
    const unsigned short* __restrict__ vb,
    const float* __restrict__ structure, const unsigned char* __restrict__ kmask,
    const float* __restrict__ wsk, const float* __restrict__ bsk,
    const float* __restrict__ wsv, const float* __restrict__ bsv,
    const float* __restrict__ wf, const float* __restrict__ bf_,
    float* __restrict__ out, float* __restrict__ fin_attn)
{
  extern __shared__ unsigned short sm[];
  unsigned short* sScore  = sm;                       // 65536 elems
  unsigned short* sQ      = sm + 65536;               //  4096 elems
  unsigned short* sSkt    = sm + 69632;               //  8192 elems
  float*          sStage0 = (float*)(sm + 77824);     //  7680 floats
  float*          sStage1 = (float*)(sm + 93184);     //  7680 floats
  unsigned short* sCtx    = (unsigned short*)sStage0; // phase-4 reuse

  const int tid  = threadIdx.x;
  const int lane = tid & 31;
  const int wave = tid >> 5;               // = head h
  const int h    = wave;
  const int lm   = lane & 15;
  const int hv   = lane >> 4;

  const int b  = blockIdx.x >> 5;
  const int q0 = (blockIdx.x & 31) * 16;

  // ---- preload wsk / wsv B-fragments (32x32, K padded from 30) -------------
  BF16x16 Bsk[2], Bsv[2];
#pragma unroll
  for (int cf = 0; cf < 2; ++cf) {
    const int n = cf * 16 + lm;
#pragma unroll
    for (int e = 0; e < 16; ++e) {
      const int s = 16 * hv + e;
      Bsk[cf].u[e] = (s < DS) ? f2bf(wsk[s * DH + n]) : (unsigned short)0;
      Bsv[cf].u[e] = (s < DS) ? f2bf(wsv[s * DH + n]) : (unsigned short)0;
    }
  }

  const float* sbase = structure + ((size_t)(b * LQ + q0) * LQ) * DS;

  // ---- TDM prologue: stage tile 0 while we also cache q rows ---------------
  if (wave == 0) tdm_load_tile(sbase, sStage0);
  for (int i = tid; i < 16 * DM; i += 256)
    sQ[i] = qb[((size_t)(b * LQ + q0) + (i >> 8)) * DM + (i & 255)];

  // ======================= Phase 1: scores ==================================
  for (int kt = 0; kt < 32; ++kt) {
    const int k0 = kt * 16;
    if (wave == 0) __builtin_amdgcn_s_wait_tensorcnt(0);   // tile kt landed
    __syncthreads();                                        // publish + free other buf
    if (wave == 0 && kt + 1 < 32)
      tdm_load_tile(sbase + (size_t)(kt + 1) * 16 * DS,
                    (kt + 1) & 1 ? (void*)sStage1 : (void*)sStage0);
    const float* stg = (kt & 1) ? sStage1 : sStage0;

    // sk tile = stage(256x30 pad 32) @ wsk(32x32) + bsk, via WMMA (4/wave)
#pragma unroll
    for (int t = 0; t < 2; ++t) {
      const int rt = 2 * wave + t;
      BF16x16 A;
#pragma unroll
      for (int e = 0; e < 16; ++e) {
        const int k = a_k(e, hv);
        A.u[e] = (k < DS) ? f2bf(stg[(rt * 16 + lm) * DS + k]) : (unsigned short)0;
      }
#pragma unroll
      for (int cf = 0; cf < 2; ++cf) {
        v8f c = {};
        c = wmma_bf16(A.v, Bsk[cf].v, c);
#pragma unroll
        for (int r = 0; r < 8; ++r) {
          const int m = r + 8 * hv, d = cf * 16 + lm;
          sSkt[(rt * 16 + m) * 32 + d] = f2bf(c[r] + bsk[d]);
        }
      }
    }
    __syncthreads();

    // score tile for this head: q(16x32) @ k^T(32x16)
    BF16x16 Aq, Bk;
#pragma unroll
    for (int e = 0; e < 16; ++e)
      Aq.u[e] = sQ[lm * DM + h * DH + a_k(e, hv)];
#pragma unroll
    for (int e = 0; e < 16; ++e) {
      const int d = 16 * hv + e;
      Bk.u[e] = kbuf[((size_t)(b * LQ) + k0 + lm) * DM + h * DH + d];
    }
    v8f sc = {};
    sc = wmma_bf16(Aq.v, Bk.v, sc);
    // + ALPHA * q . sk  (per-row structure bias, VALU), mask, store bf16
#pragma unroll
    for (int r = 0; r < 8; ++r) {
      const int m = r + 8 * hv, n = lm;
      float acc = 0.f;
#pragma unroll 8
      for (int d = 0; d < 32; ++d)
        acc += bf2f(sQ[m * DM + h * DH + d]) * bf2f(sSkt[(m * 16 + n) * 32 + d]);
      float val = sc[r] + ALPHA_C * acc;
      if (kmask[b * LQ + k0 + n]) val = -1e18f;
      sScore[h * 8192 + m * 512 + k0 + n] = f2bf(val);
    }
    // next top-of-loop barrier protects sSkt / stage reuse
  }

  // prefetch first phase-3 tile; overlaps the softmax phase
  if (wave == 0) tdm_load_tile(sbase, sStage0);

  // ======================= Phase 2: softmax (per wave/head) =================
  float inv_r = 0.f;                        // lane j (j<16) holds 1/sum of row j
  for (int r2 = 0; r2 < 16; ++r2) {
    float mx = -3.4e38f;
    for (int j = lane; j < 512; j += 32)
      mx = fmaxf(mx, bf2f(sScore[h * 8192 + r2 * 512 + j]));
#pragma unroll
    for (int o = 16; o > 0; o >>= 1) mx = fmaxf(mx, __shfl_xor(mx, o, 32));
    float sum = 0.f;
    for (int j = lane; j < 512; j += 32) {
      const float e = __expf(bf2f(sScore[h * 8192 + r2 * 512 + j]) - mx);
      sum += e;
      sScore[h * 8192 + r2 * 512 + j] = f2bf(e);
    }
#pragma unroll
    for (int o = 16; o > 0; o >>= 1) sum += __shfl_xor(sum, o, 32);
    if (lane == r2) inv_r = 1.f / sum;
  }

  // ======================= Phase 3: context =================================
  v8f C0 = {}, C1 = {};
  float cs0[8], cs1[8];
#pragma unroll
  for (int r = 0; r < 8; ++r) { cs0[r] = 0.f; cs1[r] = 0.f; }

  for (int st = 0; st < 32; ++st) {
    const int k0 = st * 16;
    if (wave == 0) __builtin_amdgcn_s_wait_tensorcnt(0);
    __syncthreads();
    if (wave == 0 && st + 1 < 32)
      tdm_load_tile(sbase + (size_t)(st + 1) * 16 * DS,
                    (st + 1) & 1 ? (void*)sStage1 : (void*)sStage0);
    const float* stg = (st & 1) ? sStage1 : sStage0;

    // sv tile via WMMA into sSkt
#pragma unroll
    for (int t = 0; t < 2; ++t) {
      const int rt = 2 * wave + t;
      BF16x16 A;
#pragma unroll
      for (int e = 0; e < 16; ++e) {
        const int k = a_k(e, hv);
        A.u[e] = (k < DS) ? f2bf(stg[(rt * 16 + lm) * DS + k]) : (unsigned short)0;
      }
#pragma unroll
      for (int cf = 0; cf < 2; ++cf) {
        v8f c = {};
        c = wmma_bf16(A.v, Bsv[cf].v, c);
#pragma unroll
        for (int r = 0; r < 8; ++r) {
          const int m = r + 8 * hv, d = cf * 16 + lm;
          sSkt[(rt * 16 + m) * 32 + d] = f2bf(c[r] + bsv[d]);
        }
      }
    }
    __syncthreads();

    // BETA * attn . sv  (per-row, VALU, unnormalized probs)
#pragma unroll
    for (int r = 0; r < 8; ++r) {
      const int m = r + 8 * hv;
      float a0 = 0.f, a1 = 0.f;
#pragma unroll 4
      for (int n = 0; n < 16; ++n) {
        const float p = bf2f(sScore[h * 8192 + m * 512 + k0 + n]);
        a0 += p * bf2f(sSkt[(m * 16 + n) * 32 + lm]);
        a1 += p * bf2f(sSkt[(m * 16 + n) * 32 + 16 + lm]);
      }
      cs0[r] += a0; cs1[r] += a1;
    }

    // every second subtile: attn(16x32) @ v(32x32), two WMMAs into ctx
    if (st & 1) {
      const int kk0 = (st - 1) * 16;
      BF16x16 Ap, Bv0, Bv1;
#pragma unroll
      for (int e = 0; e < 16; ++e)
        Ap.u[e] = sScore[h * 8192 + lm * 512 + kk0 + a_k(e, hv)];
#pragma unroll
      for (int e = 0; e < 16; ++e) {
        const int krow = 16 * hv + e;
        const size_t base = ((size_t)(b * LQ) + kk0 + krow) * DM + h * DH;
        Bv0.u[e] = vb[base + lm];
        Bv1.u[e] = vb[base + 16 + lm];
      }
      C0 = wmma_bf16(Ap.v, Bv0.v, C0);
      C1 = wmma_bf16(Ap.v, Bv1.v, C1);
    }
  }

  // ======================= Phase 4: outputs =================================
  __syncthreads();
#pragma unroll
  for (int r = 0; r < 8; ++r) {
    const int m = r + 8 * hv;
    const float inv = __shfl(inv_r, m, 32);
    sCtx[m * DM + h * DH + lm]      = f2bf((C0[r] + BETA_C * cs0[r]) * inv);
    sCtx[m * DM + h * DH + 16 + lm] = f2bf((C1[r] + BETA_C * cs1[r]) * inv);
  }
  __syncthreads();

  // head-0 normalized attention -> fin_attn (wave 0 only)
  if (wave == 0) {
    for (int i = lane; i < 16 * 512; i += 32) {
      const int m = i >> 9, k = i & 511;
      const float inv = __shfl(inv_r, m, 32);
      fin_attn[((size_t)(b * LQ) + q0 + m) * LQ + k] =
          bf2f(sScore[m * 512 + k]) * inv;
    }
  }

  // output projection: out[16,256] = ctx @ wf + bf ; wave handles 32 columns
  const int n0 = 32 * wave;
  v8f O0 = {}, O1 = {};
  for (int kk = 0; kk < 8; ++kk) {
    BF16x16 Ac, Bw0, Bw1;
#pragma unroll
    for (int e = 0; e < 16; ++e)
      Ac.u[e] = sCtx[lm * DM + kk * 32 + a_k(e, hv)];
#pragma unroll
    for (int e = 0; e < 16; ++e) {
      const int krow = kk * 32 + 16 * hv + e;
      Bw0.u[e] = f2bf(wf[(size_t)krow * DM + n0 + lm]);
      Bw1.u[e] = f2bf(wf[(size_t)krow * DM + n0 + 16 + lm]);
    }
    O0 = wmma_bf16(Ac.v, Bw0.v, O0);
    O1 = wmma_bf16(Ac.v, Bw1.v, O1);
  }
#pragma unroll
  for (int r = 0; r < 8; ++r) {
    const int m = r + 8 * hv;
    out[((size_t)(b * LQ) + q0 + m) * DM + n0 + lm]      = O0[r] + bf_[n0 + lm];
    out[((size_t)(b * LQ) + q0 + m) * DM + n0 + 16 + lm] = O1[r] + bf_[n0 + 16 + lm];
  }
}

// ---------------------------------------------------------------------------
extern "C" void kernel_launch(void* const* d_in, const int* in_sizes, int n_in,
                              void* d_out, int out_size, void* d_ws, size_t ws_size,
                              hipStream_t stream) {
  const float* key       = (const float*)d_in[0];
  const float* value     = (const float*)d_in[1];
  const float* query     = (const float*)d_in[2];
  const float* structure = (const float*)d_in[3];
  const unsigned char* kmask = (const unsigned char*)d_in[4];
  const float* wq  = (const float*)d_in[5];
  const float* bq  = (const float*)d_in[6];
  const float* wk  = (const float*)d_in[7];
  const float* bk  = (const float*)d_in[8];
  const float* wv  = (const float*)d_in[9];
  const float* bv  = (const float*)d_in[10];
  const float* wsk = (const float*)d_in[11];
  const float* bsk = (const float*)d_in[12];
  const float* wsv = (const float*)d_in[13];
  const float* bsv = (const float*)d_in[14];
  const float* wf  = (const float*)d_in[15];
  const float* bf  = (const float*)d_in[16];

  float* out      = (float*)d_out;
  float* fin_attn = out + (size_t)BS * LQ * DM;

  unsigned short* qb = (unsigned short*)d_ws;
  unsigned short* kb = qb + (size_t)BS * LQ * DM;
  unsigned short* vb = kb + (size_t)BS * LQ * DM;

  proj_qkv_kernel<<<dim3(1536), dim3(256), 0, stream>>>(
      query, key, value, wq, bq, wk, bk, wv, bv, qb, kb, vb);

  // 131072 (scores) + 8192 (q) + 16384 (skt) + 2*30720 (fp32 TDM stages)
  const size_t lds_bytes = 217088;
  attn_kernel<<<dim3(BS * 32), dim3(256), lds_bytes, stream>>>(
      qb, kb, vb, structure, kmask, wsk, bsk, wsv, bsv, wf, bf, out, fin_attn);
}
